// LambdaLayer_18751827214724
// MI455X (gfx1250) — matrix operand order repaired and checked by reading, not compiled, not run on hardware
//
#include <hip/hip_runtime.h>

typedef __attribute__((ext_vector_type(16))) __bf16 v16bf;
typedef __attribute__((ext_vector_type(8)))  __bf16 v8bf;
typedef __attribute__((ext_vector_type(8)))  float  v8f;

#define B_  32
#define D_  256
#define N_  1024   // n1*n2 = m = n
#define KD_ 16
#define H_  4
#define V_  64
#define PAD 40     // padded LDS row stride in halves (16B-aligned rows, spread banks)
#define NC  (N_ / 32)

// ------------------------------- E -> bf16, transposed per n: EbT[n][kk][m]
// makes the WMMA A-fragment a contiguous per-lane run in GLOBAL memory
__global__ void __launch_bounds__(256)
cvt_transpose_kernel(const float* __restrict__ src, __bf16* __restrict__ dst) {
    const int n = blockIdx.x;
    const float* s = src + (size_t)n * N_ * KD_;
    __bf16* d = dst + (size_t)n * KD_ * N_;
    #pragma unroll 1
    for (int kk = 0; kk < KD_; ++kk)
        for (int m = threadIdx.x; m < N_; m += 256)
            d[(size_t)kk * N_ + m] = (__bf16)s[(size_t)m * KD_ + kk];
}

// ------------------------------------------------- projections K / V / Q
__global__ void __launch_bounds__(256)
proj_kernel(const float* __restrict__ x,
            const float* __restrict__ Wq, const float* __restrict__ Wk,
            const float* __restrict__ Wv,
            float* __restrict__ Kout, float* __restrict__ Vout,
            float* __restrict__ Qout) {
    const int row = blockIdx.x;                 // b*1024 + i ; xf row is contiguous
    __shared__ float xr[D_];
    xr[threadIdx.x] = x[(size_t)row * D_ + threadIdx.x];
    __syncthreads();
    const int t = threadIdx.x;
    if (t < 144) {
        const float* w; float* dst;
        if (t < 16)      { w = Wk + t * D_;        dst = Kout + (size_t)row * KD_ + t; }
        else if (t < 80) { w = Wv + (t - 16) * D_; dst = Vout + (size_t)row * V_ + (t - 16); }
        else             { w = Wq + (t - 80) * D_; dst = Qout + (size_t)row * V_ + (t - 80); }
        float s = 0.f;
        #pragma unroll 8
        for (int j = 0; j < D_; ++j) s += xr[j] * w[j];
        *dst = s;
    }
}

// ------- values BN (stats over b,v per position m) -> bf16, TRANSPOSED
// output VbnT[b][v][m]: both the main kernel's LDS staging and the
// content-lambda kernel want v-major / m-contiguous rows.
__global__ void __launch_bounds__(256)
vbn_kernel(const float* __restrict__ Vraw, const float* __restrict__ gamma,
           const float* __restrict__ beta, __bf16* __restrict__ vbnT) {
    const int m = blockIdx.x;
    const int tid = threadIdx.x;
    __shared__ float s1[256], s2[256];
    float a = 0.f, b2 = 0.f;
    for (int idx = tid; idx < B_ * V_; idx += 256) {
        int b = idx >> 6, vv = idx & 63;
        float v = Vraw[(size_t)b * N_ * V_ + (size_t)m * V_ + vv];
        a += v; b2 += v * v;
    }
    s1[tid] = a; s2[tid] = b2; __syncthreads();
    for (int off = 128; off; off >>= 1) {
        if (tid < off) { s1[tid] += s1[tid + off]; s2[tid] += s2[tid + off]; }
        __syncthreads();
    }
    const float mean = s1[0] * (1.f / (B_ * V_));
    const float var  = s2[0] * (1.f / (B_ * V_)) - mean * mean;
    const float g    = gamma[m] * rsqrtf(var + 1e-5f);
    const float be   = beta[m];
    for (int idx = tid; idx < B_ * V_; idx += 256) {
        int b = idx >> 6, vv = idx & 63;
        float v = Vraw[(size_t)b * N_ * V_ + (size_t)m * V_ + vv];
        vbnT[(size_t)b * V_ * N_ + (size_t)vv * N_ + m] = (__bf16)((v - mean) * g + be);
    }
}

// ----------------------------------- query BN stats (per k over b,n,h)
__global__ void __launch_bounds__(256)
qstat_kernel(const float* __restrict__ Q, float* __restrict__ meanq,
             float* __restrict__ rstdq) {
    const int kk = blockIdx.x;
    const int tid = threadIdx.x;
    __shared__ float s1[256], s2[256];
    float a = 0.f, b2 = 0.f;
    for (int idx = tid; idx < B_ * N_ * H_; idx += 256) {
        int r = idx >> 2, hh = idx & 3;
        float v = Q[(size_t)r * (KD_ * H_) + kk * H_ + hh];
        a += v; b2 += v * v;
    }
    s1[tid] = a; s2[tid] = b2; __syncthreads();
    for (int off = 128; off; off >>= 1) {
        if (tid < off) { s1[tid] += s1[tid + off]; s2[tid] += s2[tid + off]; }
        __syncthreads();
    }
    if (tid == 0) {
        const float inv = 1.f / (B_ * N_ * H_);
        float mean = s1[0] * inv;
        float var  = s2[0] * inv - mean * mean;
        meanq[kk] = mean;
        rstdq[kk] = rsqrtf(var + 1e-5f);
    }
}

// ------------------------------------------ softmax over m per (b,k)
__global__ void __launch_bounds__(256)
softmax_kernel(const float* __restrict__ Keys, float* __restrict__ sk) {
    const int b = blockIdx.x >> 4, kk = blockIdx.x & 15;
    const int tid = threadIdx.x;
    const float* base = Keys + (size_t)b * N_ * KD_ + kk;
    __shared__ float red[256];
    float mx = -1e30f;
    for (int m = tid; m < N_; m += 256) mx = fmaxf(mx, base[(size_t)m * KD_]);
    red[tid] = mx; __syncthreads();
    for (int off = 128; off; off >>= 1) {
        if (tid < off) red[tid] = fmaxf(red[tid], red[tid + off]);
        __syncthreads();
    }
    const float gmx = red[0]; __syncthreads();
    float sum = 0.f;
    for (int m = tid; m < N_; m += 256) sum += __expf(base[(size_t)m * KD_] - gmx);
    red[tid] = sum; __syncthreads();
    for (int off = 128; off; off >>= 1) {
        if (tid < off) red[tid] += red[tid + off];
        __syncthreads();
    }
    const float inv = 1.f / red[0];
    float* dst = sk + (size_t)b * N_ * KD_ + kk;
    for (int m = tid; m < N_; m += 256)
        dst[(size_t)m * KD_] = __expf(base[(size_t)m * KD_] - gmx) * inv;
}

// ------------------- content lambda: bkv = sum_m sk[bmk] * vbnT[bvm]
__global__ void __launch_bounds__(64)
clam_kernel(const float* __restrict__ sk, const __bf16* __restrict__ vbnT,
            float* __restrict__ lamc) {
    const int b = blockIdx.x >> 4, kk = blockIdx.x & 15;
    const int vv = threadIdx.x;
    float acc = 0.f;
    const float*  s = sk   + (size_t)b * N_ * KD_ + kk;
    const __bf16* v = vbnT + (size_t)b * V_ * N_ + (size_t)vv * N_;
    for (int m = 0; m < N_; ++m)
        acc += s[(size_t)m * KD_] * (float)v[m];
    lamc[(size_t)b * KD_ * V_ + kk * V_ + vv] = acc;
}

// =====================================================================
// Main kernel: per (b, n) wave computes lambda_p = E[n]^T @ Vbn[b]
// via v_wmma_f32_16x16x32_bf16 (16x64 result, K=1024), adds lambda_c,
// then contracts with BN'd queries and writes the output column.
//
// A-fragments come STRAIGHT FROM GLOBAL (EbT pre-transposed so the ISA
// 7.12.2 16-bit A striping is a contiguous per-lane run), pipelined one
// chunk ahead. V (shared by all 8 waves) is pre-transposed in global
// too, so LDS staging is a pure b128 copy; B-fragments are contiguous
// ds_load_b128 runs. Double-buffered, one barrier per chunk.
// =====================================================================
__global__ void __launch_bounds__(256)
lambda_pos_kernel(const __bf16* __restrict__ EbfT, const __bf16* __restrict__ vbnT,
                  const float* __restrict__ Qraw, const float* __restrict__ lamc,
                  const float* __restrict__ meanq, const float* __restrict__ rstdq,
                  const float* __restrict__ gq, const float* __restrict__ bq,
                  float* __restrict__ out) {
    const int tid  = threadIdx.x;
    const int w    = tid >> 5;          // wave in block
    const int lane = tid & 31;
    const int half = lane >> 4;         // upper half-wave
    const int l16  = lane & 15;
    const int b    = blockIdx.x >> 7;   // 0..31
    const int n    = (blockIdx.x & 127) * 8 + w;

    __shared__ __align__(16) __bf16 VT[2][V_ * PAD];   // VT[buf][v*PAD + m], ping-pong
    __shared__ float lamLds[8][KD_ * V_];

    v8f acc[4] = {};                                   // 16x64 f32 accum

    const __bf16* vbT = vbnT + (size_t)b * V_ * N_;
    const __bf16* EbT = EbfT + (size_t)n * KD_ * N_;

    // V staging: 64 v-rows x 4 16B-segments, straight copy (already transposed)
    const int vrow = tid >> 2;          // 0..63
    const int seg  = (tid & 3) * 8;     // half offset within chunk row

    auto stageV = [&](int mc, int buf) {
        v8bf ld = *(const v8bf*)(vbT + (size_t)vrow * N_ + mc + seg);
        *(v8bf*)(&VT[buf][vrow * PAD + seg]) = ld;
    };

    union U { v16bf v; uint4 q[2]; };
    auto loadA = [&](int c) {
        // A = E^T : af[0..7] = EbT[l16][c*32 + half*8 + 0..7], af[8..15] = +16
        const __bf16* ep = EbT + (size_t)l16 * N_ + c * 32 + half * 8;
        U a;
        a.q[0] = *(const uint4*)(ep);
        a.q[1] = *(const uint4*)(ep + 16);
        return a;
    };

    stageV(0, 0);
    U a = loadA(0);

    for (int c = 0; c < NC; ++c) {
        const int cur = c & 1;
        __syncthreads();                       // staging of buf[cur] visible
        U an = a;
        if (c + 1 < NC) {
            stageV((c + 1) * 32, cur ^ 1);     // overlap next-chunk staging
            an = loadA(c + 1);                 // pipeline next A from global
            if (c + 2 < NC)
                __builtin_prefetch(EbT + (size_t)l16 * N_ + (c + 2) * 32, 0, 1);
        }
        #pragma unroll
        for (int tt = 0; tt < 4; ++tt) {
            // B : bf[0..7] = VT[tt*16+l16][half*8 + 0..7], bf[8..15] = +16
            const __bf16* vt = &VT[cur][(size_t)(tt * 16 + l16) * PAD + half * 8];
            U bu;
            bu.q[0] = *(const uint4*)(vt);
            bu.q[1] = *(const uint4*)(vt + 16);
            acc[tt] = __builtin_amdgcn_wmma_f32_16x16x32_bf16(
                false, a.v, false, bu.v, (short)0, acc[tt], false, false);
        }
        a = an;
    }
    __syncthreads();

    // lambda = lambda_p + lambda_c -> LDS  (C/D layout: M = r + 8*half, N = tt*16+l16)
    const float* lc = lamc + (size_t)b * KD_ * V_;
    #pragma unroll
    for (int tt = 0; tt < 4; ++tt)
        #pragma unroll
        for (int r = 0; r < 8; ++r) {
            const int Mr = r + half * 8;
            const int Nc = tt * 16 + l16;
            lamLds[w][Mr * V_ + Nc] = acc[tt][r] + lc[Mr * V_ + Nc];
        }
    __syncthreads();

    // out[b, h*64+v, n] = sum_k qn[b,n,k,h] * lambda[k,v]
    const int hp = lane >> 3;                       // h index for this lane
    float qn[KD_];
    const float* Qrow = Qraw + ((size_t)b * N_ + n) * (KD_ * H_);
    #pragma unroll
    for (int kk = 0; kk < KD_; ++kk)
        qn[kk] = (Qrow[kk * H_ + hp] - meanq[kk]) * rstdq[kk] * gq[kk] + bq[kk];

    float* op = out + (size_t)b * D_ * N_ + n;
    #pragma unroll
    for (int i = 0; i < 8; ++i) {
        const int vv = (lane & 7) * 8 + i;
        float s = 0.f;
        #pragma unroll
        for (int kk = 0; kk < KD_; ++kk) s += qn[kk] * lamLds[w][kk * V_ + vv];
        op[(size_t)(lane * 8 + i) * N_] = s;        // j = h*64+v = lane*8+i
    }
}

// =====================================================================
extern "C" void kernel_launch(void* const* d_in, const int* in_sizes, int n_in,
                              void* d_out, int out_size, void* d_ws, size_t ws_size,
                              hipStream_t stream) {
    const float* x  = (const float*)d_in[0];
    const float* Wq = (const float*)d_in[1];
    const float* Wk = (const float*)d_in[2];
    const float* Wv = (const float*)d_in[3];
    const float* E  = (const float*)d_in[4];
    const float* gv = (const float*)d_in[5];
    const float* bv = (const float*)d_in[6];
    const float* gq = (const float*)d_in[7];
    const float* bq = (const float*)d_in[8];
    float* out = (float*)d_out;

    char* ws = (char*)d_ws;
    size_t off = 0;
    __bf16* EbT  = (__bf16*)(ws + off); off += (size_t)N_ * N_ * KD_ * 2;   // 32 MB
    float*  Vraw = (float*) (ws + off); off += (size_t)B_ * N_ * V_ * 4;    //  8 MB
    float*  Qraw = (float*) (ws + off); off += (size_t)B_ * N_ * V_ * 4;    //  8 MB
    float*  Keys = (float*) (ws + off); off += (size_t)B_ * N_ * KD_ * 4;   //  2 MB
    float*  Sk   = (float*) (ws + off); off += (size_t)B_ * N_ * KD_ * 4;   //  2 MB
    __bf16* VbnT = (__bf16*)(ws + off); off += (size_t)B_ * N_ * V_ * 2;    //  4 MB
    float*  Lamc = (float*) (ws + off); off += (size_t)B_ * KD_ * V_ * 4;
    float*  Mq   = (float*) (ws + off); off += 256;
    float*  Rq   = (float*) (ws + off); off += 256;

    cvt_transpose_kernel<<<N_, 256, 0, stream>>>(E, EbT);
    proj_kernel<<<B_ * N_, 256, 0, stream>>>(x, Wq, Wk, Wv, Keys, Vraw, Qraw);
    vbn_kernel<<<N_, 256, 0, stream>>>(Vraw, gv, bv, VbnT);
    qstat_kernel<<<KD_, 256, 0, stream>>>(Qraw, Mq, Rq);
    softmax_kernel<<<B_ * KD_, 256, 0, stream>>>(Keys, Sk);
    clam_kernel<<<B_ * KD_, 64, 0, stream>>>(Sk, VbnT, Lamc);
    lambda_pos_kernel<<<B_ * (N_ / 8), 256, 0, stream>>>(
        EbT, VbnT, Qraw, Lamc, Mq, Rq, gq, bq, out);
}
